// TemporalCrossAttention_2628519985162
// MI455X (gfx1250) — matrix operand order repaired and checked by reading, not compiled, not run on hardware
//
#include <hip/hip_runtime.h>
#include <hip/hip_bf16.h>
#include <cstdint>

// ---------------------------------------------------------------------------
// CDNA5 WMMA types & helpers (gfx1250, wave32)
// ---------------------------------------------------------------------------
typedef __bf16 bf16_t;
typedef bf16_t v16bf __attribute__((ext_vector_type(16)));
typedef float  v8f   __attribute__((ext_vector_type(8)));

struct F8 { uint32_t u[8]; };

__device__ __forceinline__ uint16_t f2bf(float f) {
    union { float f; uint32_t u; } c; c.f = f;
    uint32_t u = c.u;
    uint32_t r = (u + 0x7FFFu + ((u >> 16) & 1u)) >> 16;   // RNE
    return (uint16_t)r;
}

// A-matrix fragment (16x32 bf16, M x K), source row-major [rows][ld] (K contiguous).
__device__ __forceinline__ v16bf load_a_frag(const uint16_t* m, int ld, int row0, int kc, int lane) {
    int half = (lane >> 4) & 1;
    const uint16_t* p = m + (size_t)(row0 + (lane & 15)) * ld + kc + half * 8;
    uint4 a = *(const uint4*)p;           // K = kc+half*8 .. +7
    uint4 b = *(const uint4*)(p + 16);    // K = kc+16+half*8 .. +7
    F8 f;
    f.u[0] = a.x; f.u[1] = a.y; f.u[2] = a.z; f.u[3] = a.w;
    f.u[4] = b.x; f.u[5] = b.y; f.u[6] = b.z; f.u[7] = b.w;
    return __builtin_bit_cast(v16bf, f);
}

// B-matrix fragment (32x16 bf16, K x N), source stored as N-major panel [n][ld].
__device__ __forceinline__ v16bf load_b_frag(const uint16_t* m, int ld, int n0, int kc, int lane) {
    int half = (lane >> 4) & 1;
    const uint16_t* p = m + (size_t)(n0 + (lane & 15)) * ld + kc + half * 8;
    uint4 a = *(const uint4*)p;
    uint4 b = *(const uint4*)(p + 16);
    F8 f;
    f.u[0] = a.x; f.u[1] = a.y; f.u[2] = a.z; f.u[3] = a.w;
    f.u[4] = b.x; f.u[5] = b.y; f.u[6] = b.z; f.u[7] = b.w;
    return __builtin_bit_cast(v16bf, f);
}

__device__ __forceinline__ v8f wmma_bf16(v16bf a, v16bf b, v8f c) {
    return __builtin_amdgcn_wmma_f32_16x16x32_bf16(false, a, false, b, (short)0, c, false, false);
}

#define V8F_ZERO {0.f,0.f,0.f,0.f,0.f,0.f,0.f,0.f}

// ---------------------------------------------------------------------------
// Async global->LDS copy (GLOBAL_LOAD_ASYNC_TO_LDS_B128, ASYNCcnt-tracked).
// VDST = per-lane LDS byte offset (hardware adds LDS_BASE), VADDR = 64-bit
// global address.  Flat address of a __shared__ object truncates to the
// addrspace(3) offset (ISA 10.2: LDS aperture uses addr[31:0]).
// ---------------------------------------------------------------------------
__device__ __forceinline__ void async_copy_b128(uint32_t lds_off, const void* gptr) {
    asm volatile("global_load_async_to_lds_b128 %0, %1, off"
                 :: "v"(lds_off), "v"(gptr) : "memory");
}
__device__ __forceinline__ void wait_async0() {
#if __has_builtin(__builtin_amdgcn_s_wait_asynccnt)
    __builtin_amdgcn_s_wait_asynccnt(0);
#else
    asm volatile("s_wait_asynccnt 0x0" ::: "memory");
#endif
}
__device__ __forceinline__ uint32_t lds_off_of(const void* p) {
    return (uint32_t)(uintptr_t)p;
}

// ---------------------------------------------------------------------------
// Problem constants
// ---------------------------------------------------------------------------
#define BATCH     2048
#define TT        16
#define QDIM      512
#define HEADS     8
#define DHEAD     64
#define INNER     512
#define ROWS      (BATCH * TT)          // 32768
#define NQKV      (3 * INNER)           // 1536
#define QKV_ELEMS ((size_t)BATCH * HEADS * TT * DHEAD)   // 16,777,216 per tensor
#define SCALE_F   0.125f                // 64^-0.5
#define RPAD      48                    // rel tables padded 33 -> 48 rows

// Workspace layout (bytes)
#define OFF_WQKV_T 0                         // [1536][512] bf16
#define OFF_WO_T   1572864                   // [512][512]  bf16
#define OFF_RK     2097152                   // [48][64]    bf16 (zero padded)
#define OFF_RV     2103296                   // [48][64]    bf16
#define OFF_QKV    2109440                   // 3 x [B*H][T][64] bf16
#define OFF_ATTN   (OFF_QKV + 3 * QKV_ELEMS * 2)  // [32768][512] bf16
// total ~136.3 MB

// ---------------------------------------------------------------------------
// Kernel 1: convert weights to bf16 panels (transposed so K is contiguous)
// ---------------------------------------------------------------------------
__global__ __launch_bounds__(256) void k_convert(
    const float* __restrict__ Wq, const float* __restrict__ Wk,
    const float* __restrict__ Wv, const float* __restrict__ Wo,
    const float* __restrict__ relk, const float* __restrict__ relv,
    uint16_t* __restrict__ wqkv_t, uint16_t* __restrict__ wo_t,
    uint16_t* __restrict__ rkp, uint16_t* __restrict__ rvp) {
    int i = blockIdx.x * 256 + threadIdx.x;
    const int N1 = NQKV * QDIM;       // 786432
    const int N2 = INNER * QDIM;      // 262144
    const int N3 = RPAD * DHEAD;      // 3072
    if (i < N1) {                     // wqkv_t[n][k] = W[k][n]
        int n = i >> 9, kk = i & 511;
        const float* W = (n < 512) ? Wq : ((n < 1024) ? Wk : Wv);
        wqkv_t[i] = f2bf(W[kk * 512 + (n & 511)]);
        return;
    }
    i -= N1;
    if (i < N2) {
        int n = i >> 9, kk = i & 511;
        wo_t[i] = f2bf(Wo[kk * 512 + n]);
        return;
    }
    i -= N2;
    if (i < N3) {
        int row = i >> 6, col = i & 63;
        rkp[i] = (row < 33) ? f2bf(relk[row * 64 + col]) : (uint16_t)0;
        return;
    }
    i -= N3;
    if (i < N3) {
        int row = i >> 6, col = i & 63;
        rvp[i] = (row < 33) ? f2bf(relv[row * 64 + col]) : (uint16_t)0;
    }
}

// ---------------------------------------------------------------------------
// Kernel 2: fused QKV projection.  [32768,512]f32 x [512,1536]bf16 -> bf16
// Block = 32 output rows staged in LDS; each wave computes a 32x64 register
// tile (2 M-frags x 4 N-frags, 8 WMMAs per K-step, A/B frags reused 4x/2x).
// 8 waves x 64 cols = 512 cols per pass, 3 passes cover N=1536.
// ---------------------------------------------------------------------------
__global__ __launch_bounds__(256) void k_qkv(
    const float* __restrict__ x, const uint16_t* __restrict__ wqkv_t,
    uint16_t* __restrict__ qkv) {
    __shared__ __align__(16) uint16_t A[32 * 512];
    int m0 = blockIdx.x * 32;
    int tid = threadIdx.x;
    for (int i = tid; i < 32 * 512; i += 256) {
        int r = i >> 9, c = i & 511;
        A[i] = f2bf(x[(size_t)(m0 + r) * 512 + c]);
    }
    __syncthreads();
    int wave = tid >> 5, lane = tid & 31;
    int half = lane >> 4;
    for (int pass = 0; pass < 3; pass++) {
        int n0 = pass * 512 + wave * 64;
        v8f acc[2][4];
        {
            v8f z = V8F_ZERO;
            for (int mf = 0; mf < 2; mf++)
                for (int j = 0; j < 4; j++) acc[mf][j] = z;
        }
        for (int kc = 0; kc < 512; kc += 32) {
            v16bf a0 = load_a_frag(A, 512, 0, kc, lane);
            v16bf a1 = load_a_frag(A, 512, 16, kc, lane);
            for (int j = 0; j < 4; j++) {
                v16bf b = load_b_frag(wqkv_t, 512, n0 + j * 16, kc, lane);
                acc[0][j] = wmma_bf16(a0, b, acc[0][j]);
                acc[1][j] = wmma_bf16(a1, b, acc[1][j]);
            }
        }
        for (int j = 0; j < 4; j++) {
            int n = n0 + j * 16 + (lane & 15);
            int which = n >> 9;           // 0=Q 1=K 2=V
            int nn = n & 511;
            int h = nn >> 6, d = nn & 63;
            for (int mf = 0; mf < 2; mf++)
                for (int r = 0; r < 8; r++) {
                    int m = m0 + mf * 16 + r + half * 8;
                    int bi = m >> 4, t = m & 15;
                    qkv[(size_t)which * QKV_ELEMS +
                        (((size_t)(bi * HEADS + h)) * TT + t) * DHEAD + d] =
                        f2bf(acc[mf][j][r]);
                }
        }
    }
}

// ---------------------------------------------------------------------------
// Kernel 3: fused attention per (b,h).  One wave32 per block.
//   scores = q @ [K | Rk]^T  (16x64 via 8 WMMAs)
//   softmax with rel gather (col s-t+32), no clip needed since T<=MAX_REL
//   out    = [attn | attn_shifted] @ [V ; Rv]  (16x64 via 8 WMMAs)
// K|Rk panel staged with async global->LDS DMA; V/Rv transposed via VALU.
// ---------------------------------------------------------------------------
__global__ __launch_bounds__(32) void k_attn(
    const uint16_t* __restrict__ qkv, const uint16_t* __restrict__ rkp,
    const uint16_t* __restrict__ rvp, uint16_t* __restrict__ attnout) {
    __shared__ __align__(16) uint16_t B1[64 * 64];  // [n][d]: rows 0-15 = K, 16-63 = Rk
    __shared__ __align__(16) uint16_t B2[64 * 64];  // [d][r]: cols 0-15 = V^T, 16-63 = Rv^T
    __shared__ __align__(16) uint16_t Ap[16 * 64];  // [t][r]: attn | shifted attn
    __shared__ float S[16 * 64];                    // raw score tile
    int l = threadIdx.x;
    int bh = blockIdx.x;                            // b*8 + h
    const uint16_t* q  = qkv + (size_t)bh * (TT * DHEAD);
    const uint16_t* kk = qkv + QKV_ELEMS + (size_t)bh * (TT * DHEAD);
    const uint16_t* vv = qkv + 2 * QKV_ELEMS + (size_t)bh * (TT * DHEAD);

    // straight copies -> async DMA (ASYNCcnt)
    uint32_t b1off = lds_off_of(&B1[0]);
    for (int i = l; i < 128; i += 32)               // K rows: 2 KB
        async_copy_b128(b1off + i * 16, (const uint8_t*)kk + i * 16);
    for (int i = l; i < 384; i += 32)               // Rk rows: 6 KB
        async_copy_b128(b1off + 2048 + i * 16, (const uint8_t*)rkp + i * 16);
    // transposed copies -> VALU
    for (int i = l; i < 1024; i += 32) { int s = i >> 6, d = i & 63; B2[d * 64 + s] = vv[i]; }
    for (int i = l; i < 3072; i += 32) { int j = i >> 6, d = i & 63; B2[d * 64 + 16 + j] = rvp[i]; }
    wait_async0();
    __syncthreads();

    // scores: 16 x 64 (cols 0-15: q.k^T ; cols 16-63: q.Rk^T)
    v8f acc[4] = { V8F_ZERO, V8F_ZERO, V8F_ZERO, V8F_ZERO };
    for (int kc = 0; kc < 64; kc += 32) {
        v16bf a = load_a_frag(q, 64, 0, kc, l);
        for (int nt = 0; nt < 4; nt++) {
            v16bf b = load_b_frag(B1, 64, nt * 16, kc, l);
            acc[nt] = wmma_bf16(a, b, acc[nt]);
        }
    }
    int half = l >> 4, col = l & 15;
    for (int nt = 0; nt < 4; nt++)
        for (int r = 0; r < 8; r++)
            S[(r + half * 8) * 64 + nt * 16 + col] = acc[nt][r];
    __syncthreads();

    // softmax (lanes 0..15, one row each); build shifted A-panel
    if (l < 16) {
        int t = l;
        float e[16]; float mx = -1e30f;
        for (int s = 0; s < 16; s++) {
            float v = SCALE_F * (S[t * 64 + s] + S[t * 64 + (s - t + 32)]);
            e[s] = v; mx = fmaxf(mx, v);
        }
        float sum = 0.f;
        for (int s = 0; s < 16; s++) { e[s] = __expf(e[s] - mx); sum += e[s]; }
        float inv = 1.f / sum;
        for (int r = 0; r < 16; r++) Ap[t * 64 + r] = f2bf(e[r] * inv);
        for (int r = 16; r < 64; r++) {
            int s = r + t - 32;                     // j = r-16 ; s = j + t - 16
            Ap[t * 64 + r] = (s >= 0 && s < 16) ? f2bf(e[s] * inv) : (uint16_t)0;
        }
    }
    __syncthreads();

    // out = Ap(16x64) @ [V ; Rv](64x64)
    v8f oc[4] = { V8F_ZERO, V8F_ZERO, V8F_ZERO, V8F_ZERO };
    for (int kc = 0; kc < 64; kc += 32) {
        v16bf a = load_a_frag(Ap, 64, 0, kc, l);
        for (int nt = 0; nt < 4; nt++) {
            v16bf b = load_b_frag(B2, 64, nt * 16, kc, l);
            oc[nt] = wmma_bf16(a, b, oc[nt]);
        }
    }
    int b_ = bh >> 3, h = bh & 7;
    for (int nt = 0; nt < 4; nt++)
        for (int r = 0; r < 8; r++) {
            int t = r + half * 8, d = nt * 16 + col;
            attnout[((size_t)(b_ * TT + t)) * INNER + h * DHEAD + d] = f2bf(oc[nt][r]);
        }
}

// ---------------------------------------------------------------------------
// Kernel 4: output projection + bias. [32768,512]bf16 x [512,512]bf16 -> f32
// A tile staged via async global->LDS DMA; 32x64 register tile per wave.
// ---------------------------------------------------------------------------
__global__ __launch_bounds__(256) void k_out(
    const uint16_t* __restrict__ attnout, const uint16_t* __restrict__ wo_t,
    const float* __restrict__ bo, float* __restrict__ out) {
    __shared__ __align__(16) uint16_t A[32 * 512];
    int m0 = blockIdx.x * 32;
    int tid = threadIdx.x;
    uint32_t aoff = lds_off_of(&A[0]);
    const uint8_t* src = (const uint8_t*)(attnout + (size_t)m0 * 512);
    for (int i = tid; i < 2048; i += 256)           // 32 KB in b128 chunks
        async_copy_b128(aoff + i * 16, src + i * 16);
    wait_async0();
    __syncthreads();

    int wave = tid >> 5, lane = tid & 31;
    int half = lane >> 4;
    int n0 = wave * 64;
    v8f acc[2][4];
    {
        v8f z = V8F_ZERO;
        for (int mf = 0; mf < 2; mf++)
            for (int j = 0; j < 4; j++) acc[mf][j] = z;
    }
    for (int kc = 0; kc < 512; kc += 32) {
        v16bf a0 = load_a_frag(A, 512, 0, kc, lane);
        v16bf a1 = load_a_frag(A, 512, 16, kc, lane);
        for (int j = 0; j < 4; j++) {
            v16bf b = load_b_frag(wo_t, 512, n0 + j * 16, kc, lane);
            acc[0][j] = wmma_bf16(a0, b, acc[0][j]);
            acc[1][j] = wmma_bf16(a1, b, acc[1][j]);
        }
    }
    for (int j = 0; j < 4; j++) {
        int n = n0 + j * 16 + (lane & 15);
        float bias = bo[n];
        for (int mf = 0; mf < 2; mf++)
            for (int r = 0; r < 8; r++) {
                int m = m0 + mf * 16 + r + half * 8;
                out[(size_t)m * 512 + n] = acc[mf][j][r] + bias;
            }
    }
}

// ---------------------------------------------------------------------------
// Launcher
// ---------------------------------------------------------------------------
extern "C" void kernel_launch(void* const* d_in, const int* in_sizes, int n_in,
                              void* d_out, int out_size, void* d_ws, size_t ws_size,
                              hipStream_t stream) {
    const float* x    = (const float*)d_in[0];
    const float* Wq   = (const float*)d_in[1];
    const float* Wk   = (const float*)d_in[2];
    const float* Wv   = (const float*)d_in[3];
    const float* Wo   = (const float*)d_in[4];
    const float* bo   = (const float*)d_in[5];
    const float* relk = (const float*)d_in[6];
    const float* relv = (const float*)d_in[7];
    float* out = (float*)d_out;

    uint8_t* ws = (uint8_t*)d_ws;
    uint16_t* wqkv_t  = (uint16_t*)(ws + OFF_WQKV_T);
    uint16_t* wo_t    = (uint16_t*)(ws + OFF_WO_T);
    uint16_t* rkp     = (uint16_t*)(ws + OFF_RK);
    uint16_t* rvp     = (uint16_t*)(ws + OFF_RV);
    uint16_t* qkv     = (uint16_t*)(ws + OFF_QKV);
    uint16_t* attnbuf = (uint16_t*)(ws + OFF_ATTN);

    const int conv_elems = NQKV * QDIM + INNER * QDIM + 2 * RPAD * DHEAD;  // 1,054,720
    k_convert<<<(conv_elems + 255) / 256, 256, 0, stream>>>(
        Wq, Wk, Wv, Wo, relk, relv, wqkv_t, wo_t, rkp, rvp);

    k_qkv<<<ROWS / 32, 256, 0, stream>>>(x, wqkv_t, qkv);

    k_attn<<<BATCH * HEADS, 32, 0, stream>>>(qkv, rkp, rvp, attnbuf);

    k_out<<<ROWS / 32, 256, 0, stream>>>(attnbuf, wo_t, bo, out);
}
